// CustomLSTM_28501402976560
// MI455X (gfx1250) — compile-verified, matrix operand order checked by reading, not run on hardware
//
#include <hip/hip_runtime.h>
#include <hip/hip_bf16.h>
#include <stdint.h>

// Problem sizes (compile-time constants from the reference)
#define B_  64
#define S_  512
#define I_  512
#define H_  1024
#define O_  512
#define K_  (H_ + I_)      // 1536
#define NWG 32             // persistent workgroups for the recurrent kernel
#define TPB 256            // 8 wave32s per workgroup

typedef __attribute__((ext_vector_type(16))) __bf16 v16bf;
typedef __attribute__((ext_vector_type(8)))  float  v8f;

union FragBF { v16bf v; uint4 q[2]; };

// LDS A-tile pitch in bf16 elements (128 data + 8 pad -> 272B rows, 16B aligned)
#define APITCH 136

// ---------------------------------------------------------------------------
// CDNA5 async global->LDS staging (ASYNCcnt-tracked, no VGPR round trip).
// Probe established: first param of the builtin is an AS(1) int4* pointer.
// ---------------------------------------------------------------------------
#if __has_builtin(__builtin_amdgcn_global_load_async_to_lds_b128)
#define HAVE_ASYNC_LDS 1
typedef int v4i_ __attribute__((ext_vector_type(4)));
typedef __attribute__((address_space(1))) v4i_ g_v4i;
typedef __attribute__((address_space(3))) v4i_ l_v4i;
#else
#define HAVE_ASYNC_LDS 0
#endif

__device__ inline void stage64B(const unsigned short* __restrict__ gsrc,
                                unsigned short* __restrict__ ldst) {
#if HAVE_ASYNC_LDS
    g_v4i* g = (g_v4i*)gsrc;
    l_v4i* l = (l_v4i*)ldst;
    __builtin_amdgcn_global_load_async_to_lds_b128(g,     l,     0, 0);
    __builtin_amdgcn_global_load_async_to_lds_b128(g + 1, l + 1, 0, 0);
    __builtin_amdgcn_global_load_async_to_lds_b128(g + 2, l + 2, 0, 0);
    __builtin_amdgcn_global_load_async_to_lds_b128(g + 3, l + 3, 0, 0);
#else
    const uint4* s4 = (const uint4*)gsrc;
    uint4* d4 = (uint4*)ldst;
    uint4 t0 = s4[0], t1 = s4[1], t2 = s4[2], t3 = s4[3];
    d4[0] = t0; d4[1] = t1; d4[2] = t2; d4[3] = t3;
#endif
}

__device__ inline void stage_wait() {
#if HAVE_ASYNC_LDS
#if __has_builtin(__builtin_amdgcn_s_wait_asynccnt)
    __builtin_amdgcn_s_wait_asynccnt(0);
#else
    asm volatile("s_wait_asynccnt 0x0" ::: "memory");
#endif
#endif
}

// ---------------------------------------------------------------------------
// A-fragment loader from LDS (16x32 bf16 tile, ISA 7.12.2 layout):
// lanes 0-15 hold row M=lane: elems 0..7 = K k0..k0+7, elems 8..15 = K k0+16..23
// lanes 16-31 hold row M=lane-16: K ranges shifted by +8.
// ---------------------------------------------------------------------------
__device__ inline v16bf lds_a_frag(const unsigned short* sA, int row, int k0, int lane) {
    const int kb = k0 + ((lane >> 4) << 3);
    const uint4* p = (const uint4*)(sA + (size_t)row * APITCH + kb);
    FragBF f;
    f.q[0] = p[0];      // K kb .. kb+7
    f.q[1] = p[2];      // K kb+16 .. kb+23   (+32 bytes)
    return f.v;
}

// B-fragment loader from global bf16 weights stored row-major [N][K]
// (mirror layout: lanes 0-15 -> column N=lane, same K striping as A).
__device__ inline v16bf glb_b_frag(const unsigned short* Wbase, size_t rowStrideElems,
                                   int ncol, int kg, int lane) {
    const int n  = (lane & 15);
    const int kb = kg + ((lane >> 4) << 3);
    const uint4* p = (const uint4*)(Wbase + (size_t)(ncol + n) * rowStrideElems + kb);
    FragBF f;
    f.q[0] = p[0];
    f.q[1] = p[2];
    return f.v;
}

// ---------------------------------------------------------------------------
// fp32 -> bf16 conversion (grid-stride)
// ---------------------------------------------------------------------------
__global__ void cvt_f32_bf16(const float* __restrict__ src,
                             __hip_bfloat16* __restrict__ dst, size_t n) {
    size_t i = (size_t)blockIdx.x * blockDim.x + threadIdx.x;
    size_t stride = (size_t)gridDim.x * blockDim.x;
    for (; i < n; i += stride) dst[i] = __float2bfloat16(src[i]);
}

__global__ void zero_u32(unsigned int* __restrict__ p, size_t n) {
    size_t i = (size_t)blockIdx.x * blockDim.x + threadIdx.x;
    size_t stride = (size_t)gridDim.x * blockDim.x;
    for (; i < n; i += stride) p[i] = 0u;
}

// ---------------------------------------------------------------------------
// Persistent recurrent kernel: 32 WGs x 256 threads, one WG per 32 hidden units.
// Per step: g = [h,x_t] @ W4^T  via v_wmma_f32_16x16x32_bf16, then gate math.
// ---------------------------------------------------------------------------
__global__ __launch_bounds__(TPB)
void lstm_kernel(const __hip_bfloat16* __restrict__ xb,     // [B][S][I] bf16
                 const __hip_bfloat16* __restrict__ W4,     // [4H][K]  bf16, gate-major
                 const float* __restrict__ bfv, const float* __restrict__ biv,
                 const float* __restrict__ bov, const float* __restrict__ bcv,
                 __hip_bfloat16* __restrict__ hbuf,         // [2][B*H] bf16 (double buffer)
                 float* __restrict__ cst,                   // [B*H] fp32 cell state
                 __hip_bfloat16* __restrict__ hseq,         // [B][S][H] bf16
                 unsigned int* __restrict__ bar)            // grid barrier counter
{
    __shared__ unsigned short sA[64 * APITCH];   // 64 x 128 bf16 activation chunk
    __shared__ float sG[4][64][33];              // gate exchange: [gate][batch][hid]

    const int tid  = threadIdx.x;
    const int wave = tid >> 5;
    const int lane = tid & 31;
    const int wg   = blockIdx.x;          // 0..31
    const int gate = wave >> 1;           // 0..3 (f,i,o,c)
    const int sub  = wave & 1;            // 16-col subtile
    const int hu0  = wg * 32;             // hidden base of this WG
    const int col0 = hu0 + sub * 16;      // this wave's hidden column base
    const int grow0 = gate * H_ + col0;   // row base in W4

    // cooperative stage mapping: each thread moves one 64B segment
    const int ldRow = tid >> 2;           // batch row 0..63
    const int ldSeg = (tid & 3) * 32;     // element offset within 128-wide chunk

    for (int s = 0; s < S_; ++s) {
        const int par = s & 1;
        const __hip_bfloat16* hprev = hbuf + (size_t)par * (B_ * H_);
        __hip_bfloat16*       hnext = hbuf + (size_t)(par ^ 1) * (B_ * H_);

        v8f acc[4] = {};

        // K = 1536 in 12 LDS-staged chunks of 128
        for (int kc = 0; kc < K_ / 128; ++kc) {
            const int kb0 = kc * 128;
            __syncthreads();
            {
                const int gc = kb0 + ldSeg;
                const unsigned short* src;
                if (gc < H_)
                    src = (const unsigned short*)hprev + (size_t)ldRow * H_ + gc;
                else
                    src = (const unsigned short*)xb +
                          ((size_t)ldRow * S_ + s) * I_ + (gc - H_);
                stage64B(src, sA + (size_t)ldRow * APITCH + ldSeg);
            }
            stage_wait();
            __syncthreads();

            // Software-pipelined B fragments: prefetch ks+1 while computing ks.
            v16bf bcur = glb_b_frag((const unsigned short*)W4, (size_t)K_,
                                    grow0, kb0, lane);
            #pragma unroll
            for (int ks = 0; ks < 4; ++ks) {
                v16bf bnext;
                if (ks < 3)
                    bnext = glb_b_frag((const unsigned short*)W4, (size_t)K_,
                                       grow0, kb0 + (ks + 1) * 32, lane);
                const int k0l = ks * 32;
                #pragma unroll
                for (int mt = 0; mt < 4; ++mt) {
                    const v16bf afrag = lds_a_frag(sA, mt * 16 + (lane & 15), k0l, lane);
                    acc[mt] = __builtin_amdgcn_wmma_f32_16x16x32_bf16(
                        false, afrag, false, bcur, (short)0, acc[mt], false, false);
                }
                bcur = bnext;
            }
        }

        // Scatter accumulators to LDS gate buffer (C/D layout: VGPR v -> M=v / v+8)
        #pragma unroll
        for (int mt = 0; mt < 4; ++mt) {
            const int rbase = mt * 16 + ((lane < 16) ? 0 : 8);
            const int n = lane & 15;
            #pragma unroll
            for (int v = 0; v < 8; ++v)
                sG[gate][rbase + v][sub * 16 + n] = acc[mt][v];
        }
        __syncthreads();

        // Elementwise LSTM cell update (fp32): 2048 values / 256 threads
        #pragma unroll
        for (int q = 0; q < 8; ++q) {
            const int lin = q * TPB + tid;
            const int row = lin >> 5;      // batch
            const int col = lin & 31;      // hidden within WG
            const int hu  = hu0 + col;
            const float fg = sG[0][row][col] + bfv[hu];
            const float ig = sG[1][row][col] + biv[hu];
            const float og = sG[2][row][col] + bov[hu];
            const float cg = sG[3][row][col] + bcv[hu];
            const size_t idx = (size_t)row * H_ + hu;
            const float c_old = cst[idx];
            const float sf = 1.f / (1.f + __expf(-fg));
            const float si = 1.f / (1.f + __expf(-ig));
            const float so = 1.f / (1.f + __expf(-og));
            const float cn = sf * c_old + si * tanhf(cg);
            const float hn = so * tanhf(cn);
            cst[idx] = cn;
            const __hip_bfloat16 hb = __float2bfloat16(hn);
            hnext[idx] = hb;
            hseq[((size_t)row * S_ + s) * H_ + hu] = hb;
        }

        // Device-wide step barrier (monotonic counter; release/acquire)
        __threadfence();
        __syncthreads();
        if (tid == 0) {
            __atomic_fetch_add(bar, 1u, __ATOMIC_RELEASE);
            const unsigned int target = (unsigned int)(NWG * (s + 1));
            while (__atomic_load_n(bar, __ATOMIC_ACQUIRE) < target)
                __builtin_amdgcn_s_sleep(1);
        }
        __syncthreads();
        __threadfence();
    }
}

// ---------------------------------------------------------------------------
// Final FC: out[B*S, O] = hseq[B*S, H] @ W_fc^T + b_fc    (bf16 WMMA, f32 out)
// Grid: 512 row-blocks x 4 col-blocks; WG = 64 rows x 128 cols.
// ---------------------------------------------------------------------------
__global__ __launch_bounds__(TPB)
void fc_kernel(const __hip_bfloat16* __restrict__ hseq,
               const __hip_bfloat16* __restrict__ wfc,     // [O][H] bf16
               const float* __restrict__ bfc,
               float* __restrict__ out)                    // [B*S][O] fp32
{
    __shared__ unsigned short sA[64 * APITCH];

    const int tid  = threadIdx.x;
    const int wave = tid >> 5;
    const int lane = tid & 31;
    const int rb = blockIdx.x >> 2;       // 0..511
    const int cb = blockIdx.x & 3;        // 0..3
    const int r0 = rb * 64;
    const int c0 = cb * 128 + wave * 16;

    const int ldRow = tid >> 2;
    const int ldSeg = (tid & 3) * 32;

    v8f acc[4] = {};
    for (int kc = 0; kc < H_ / 128; ++kc) {
        const int kb0 = kc * 128;
        __syncthreads();
        stage64B((const unsigned short*)hseq + (size_t)(r0 + ldRow) * H_ + kb0 + ldSeg,
                 sA + (size_t)ldRow * APITCH + ldSeg);
        stage_wait();
        __syncthreads();

        v16bf bcur = glb_b_frag((const unsigned short*)wfc, (size_t)H_,
                                c0, kb0, lane);
        #pragma unroll
        for (int ks = 0; ks < 4; ++ks) {
            v16bf bnext;
            if (ks < 3)
                bnext = glb_b_frag((const unsigned short*)wfc, (size_t)H_,
                                   c0, kb0 + (ks + 1) * 32, lane);
            const int k0l = ks * 32;
            #pragma unroll
            for (int mt = 0; mt < 4; ++mt) {
                const v16bf afrag = lds_a_frag(sA, mt * 16 + (lane & 15), k0l, lane);
                acc[mt] = __builtin_amdgcn_wmma_f32_16x16x32_bf16(
                    false, afrag, false, bcur, (short)0, acc[mt], false, false);
            }
            bcur = bnext;
        }
    }

    #pragma unroll
    for (int mt = 0; mt < 4; ++mt) {
        const int rbase = mt * 16 + ((lane < 16) ? 0 : 8);
        const int n = lane & 15;
        const int oc = c0 + n;
        const float bias = bfc[oc];
        #pragma unroll
        for (int v = 0; v < 8; ++v)
            out[(size_t)(r0 + rbase + v) * O_ + oc] = acc[mt][v] + bias;
    }
}

// Final (h, c) outputs: h from bf16 buffer (parity 0 after 512 steps), c fp32.
__global__ void tail_kernel(const __hip_bfloat16* __restrict__ hfin,
                            const float* __restrict__ cst,
                            float* __restrict__ out_h, float* __restrict__ out_c) {
    const int i = blockIdx.x * blockDim.x + threadIdx.x;
    if (i < B_ * H_) {
        out_h[i] = __bfloat162float(hfin[i]);
        out_c[i] = cst[i];
    }
}

// ---------------------------------------------------------------------------
extern "C" void kernel_launch(void* const* d_in, const int* in_sizes, int n_in,
                              void* d_out, int out_size, void* d_ws, size_t ws_size,
                              hipStream_t stream) {
    const float* x   = (const float*)d_in[0];
    const float* Wf  = (const float*)d_in[1];
    const float* Wi  = (const float*)d_in[2];
    const float* Wo  = (const float*)d_in[3];
    const float* Wc  = (const float*)d_in[4];
    const float* bfv = (const float*)d_in[5];
    const float* biv = (const float*)d_in[6];
    const float* bov = (const float*)d_in[7];
    const float* bcv = (const float*)d_in[8];
    const float* Wfc = (const float*)d_in[9];
    const float* bfc = (const float*)d_in[10];

    char* ws = (char*)d_ws;
    size_t off = 0;
    auto alloc = [&](size_t bytes) {
        size_t o = off;
        off += (bytes + 255) & ~(size_t)255;
        return o;
    };
    const size_t oW4   = alloc((size_t)4 * H_ * K_ * 2);        // 12.6 MB
    const size_t oWFC  = alloc((size_t)O_ * H_ * 2);            //  1.0 MB
    const size_t oXB   = alloc((size_t)B_ * S_ * I_ * 2);       // 33.6 MB
    const size_t oHSEQ = alloc((size_t)B_ * S_ * H_ * 2);       // 67.1 MB
    const size_t oHBUF = alloc((size_t)2 * B_ * H_ * 2);        // 256 KB
    const size_t oCST  = alloc((size_t)B_ * H_ * 4);            // 256 KB
    const size_t oBAR  = alloc(256);

    __hip_bfloat16* W4   = (__hip_bfloat16*)(ws + oW4);
    __hip_bfloat16* WFC  = (__hip_bfloat16*)(ws + oWFC);
    __hip_bfloat16* XB   = (__hip_bfloat16*)(ws + oXB);
    __hip_bfloat16* HSEQ = (__hip_bfloat16*)(ws + oHSEQ);
    __hip_bfloat16* HBUF = (__hip_bfloat16*)(ws + oHBUF);
    float*          CST  = (float*)(ws + oCST);
    unsigned int*   BAR  = (unsigned int*)(ws + oBAR);

    // fp32 -> bf16 conversions (weights gate-major, activations)
    const size_t nWg = (size_t)H_ * K_;
    cvt_f32_bf16<<<1024, 256, 0, stream>>>(Wf, W4 + 0 * nWg, nWg);
    cvt_f32_bf16<<<1024, 256, 0, stream>>>(Wi, W4 + 1 * nWg, nWg);
    cvt_f32_bf16<<<1024, 256, 0, stream>>>(Wo, W4 + 2 * nWg, nWg);
    cvt_f32_bf16<<<1024, 256, 0, stream>>>(Wc, W4 + 3 * nWg, nWg);
    cvt_f32_bf16<<<1024, 256, 0, stream>>>(Wfc, WFC, (size_t)O_ * H_);
    cvt_f32_bf16<<<2048, 256, 0, stream>>>(x, XB, (size_t)B_ * S_ * I_);

    // Zero h0/c0 and the barrier counter (contiguous span in ws)
    const size_t zbytes = (oBAR + 256) - oHBUF;
    zero_u32<<<512, 256, 0, stream>>>((unsigned int*)(ws + oHBUF), zbytes / 4);

    // Recurrent loop (persistent grid, one device barrier per timestep)
    lstm_kernel<<<NWG, TPB, 0, stream>>>(XB, W4, bfv, biv, bov, bcv,
                                         HBUF, CST, HSEQ, BAR);

    // Final projection + tail states
    fc_kernel<<<(B_ * S_ / 64) * (O_ / 128), TPB, 0, stream>>>(
        HSEQ, WFC, bfc, (float*)d_out);

    float* out_h = (float*)d_out + (size_t)B_ * S_ * O_;
    float* out_c = out_h + (size_t)B_ * H_;
    tail_kernel<<<(B_ * H_ + 255) / 256, 256, 0, stream>>>(HBUF, CST, out_h, out_c);
}
